// STULayer_566935683421
// MI455X (gfx1250) — matrix operand order repaired and checked by reading, not compiled, not run on hardware
//
#include <hip/hip_runtime.h>
#include <hip/hip_bf16.h>
#include <stdint.h>

// Problem constants
#define B_   8
#define S_   2048
#define E_   512
#define H_   8
#define AD_  64
#define HD_  64
#define F_   2048   // uvqk out dim = (HD*2+AD*2)*H
#define TH_  512    // H*HD
#define YK_  1536   // HD*H*2 + E
#define M_   16384  // B*S tokens

typedef __attribute__((ext_vector_type(16))) __bf16 v16bf;
typedef __attribute__((ext_vector_type(8)))  float  v8f;

// 16-byte chunk type (avoid HIP uint4 ctor issues inside unions)
struct __align__(16) U16B { unsigned int w[4]; };

union BFrag {
    v16bf v;
    U16B  q[2];
};

__device__ __forceinline__ unsigned short f2bf(float f) {
    unsigned int u = __builtin_bit_cast(unsigned int, f);
    unsigned int r = u + 0x7FFFu + ((u >> 16) & 1u);   // round-to-nearest-even
    return (unsigned short)(r >> 16);
}
__device__ __forceinline__ float bf2f(unsigned short h) {
    unsigned int u = ((unsigned int)h) << 16;
    return __builtin_bit_cast(float, u);
}
__device__ __forceinline__ v8f wmma_bf16(const BFrag& a, const BFrag& b, v8f c) {
    return __builtin_amdgcn_wmma_f32_16x16x32_bf16(false, a.v, false, b.v,
                                                   (short)0, c, false, false);
}
__device__ __forceinline__ float silu(float x) {
    return x / (1.0f + __expf(-x));
}

// Async global->LDS 16B move, per lane (GV mode). lds_addr = low 32 bits of the
// flat shared address (HW truncates flat LDS addresses to addr[31:0]).
__device__ __forceinline__ void async_b128(unsigned lds_addr, const void* gaddr) {
    asm volatile("global_load_async_to_lds_b128 %0, %1, off"
                 :: "v"(lds_addr), "v"(gaddr) : "memory");
}
__device__ __forceinline__ void wait_async0() {
    asm volatile("s_wait_asynccnt 0x0" ::: "memory");
}

// Issue async loads for one 128x32 A tile + 128x32 B tile (bf16) into buffer.
// 256 threads * 2 chunks each per operand; each lane moves 16B per instr.
__device__ __forceinline__ void tile_async_load(
        const unsigned short* __restrict__ Abf,
        const unsigned short* __restrict__ Bt,
        int K, int bm0, int bn0, int k0, int tid,
        unsigned asBase, unsigned bsBase, int buf) {
#pragma unroll
    for (int i = 0; i < 2; ++i) {
        int idx = tid * 2 + i;               // 0..511
        int row = idx >> 2, seg = idx & 3;
        unsigned loff = (unsigned)(buf * (128 * 32 * 2) + (row * 32 + seg * 8) * 2);
        const unsigned short* ga = &Abf[(size_t)(bm0 + row) * K + k0 + seg * 8];
        const unsigned short* gb = &Bt [(size_t)(bn0 + row) * K + k0 + seg * 8];
        async_b128(asBase + loff, ga);
        async_b128(bsBase + loff, gb);
    }
}

// ---------------------------------------------------------------------------
// Weight convert + transpose: W[E][F] (f32) -> Wt[F][E] (bf16)
// ---------------------------------------------------------------------------
__global__ __launch_bounds__(256) void convert_wu_kernel(const float* __restrict__ W,
                                                         unsigned short* __restrict__ Wt) {
    int idx = blockIdx.x * 256 + threadIdx.x;   // over E_*F_
    int k = idx >> 11;          // / F_
    int n = idx & (F_ - 1);
    Wt[(size_t)n * E_ + k] = f2bf(W[idx]);
}
// W[YK][E] (f32) -> Wt[E][YK] (bf16)
__global__ __launch_bounds__(256) void convert_wo_kernel(const float* __restrict__ W,
                                                         unsigned short* __restrict__ Wt) {
    int idx = blockIdx.x * 256 + threadIdx.x;   // over YK_*E_
    int k = idx >> 9;           // / E_
    int n = idx & (E_ - 1);
    Wt[(size_t)n * YK_ + k] = f2bf(W[idx]);
}

// ---------------------------------------------------------------------------
// LN1: one wave per token; write normalized bf16 activations
// ---------------------------------------------------------------------------
__global__ __launch_bounds__(256) void ln1_kernel(const float* __restrict__ x,
                                                  const float* __restrict__ g,
                                                  const float* __restrict__ be,
                                                  unsigned short* __restrict__ Xbf) {
    int w = threadIdx.x >> 5, lane = threadIdx.x & 31;
    int t = blockIdx.x * 8 + w;
    const float* xr = x + (size_t)t * E_;
    float vals[16];
    float s = 0.f, s2 = 0.f;
#pragma unroll
    for (int j = 0; j < 16; ++j) {
        float v = xr[lane + 32 * j];
        vals[j] = v; s += v; s2 += v * v;
    }
#pragma unroll
    for (int off = 16; off >= 1; off >>= 1) {
        s  += __shfl_xor(s,  off, 32);
        s2 += __shfl_xor(s2, off, 32);
    }
    float mean = s * (1.0f / E_);
    float var  = s2 * (1.0f / E_) - mean * mean;
    float rinv = rsqrtf(var + 1e-6f);
#pragma unroll
    for (int j = 0; j < 16; ++j) {
        int i = lane + 32 * j;
        float nv = (vals[j] - mean) * rinv * g[i] + be[i];
        Xbf[(size_t)t * E_ + i] = f2bf(nv);
    }
}

// ---------------------------------------------------------------------------
// GEMM1: uvqk = Xbf(16384x512) @ Wu(512x2048) + bias, fused split epilogue
//   cols [0,512)    -> u  = silu(.)            -> u_bf [M][512]
//   cols [512,1024) -> v  -> vt_bf [b][h][d][S]   (d-major for attention B)
//   cols [1024,1536)-> q  -> q_bf  [b][h][S][AD]
//   cols [1536,2048)-> k  -> k_bf  [b][h][S][AD]
// Double-buffered LDS fed by async global->LDS loads (ASYNCcnt).
// ---------------------------------------------------------------------------
__global__ __launch_bounds__(256) void gemm_uvqk_kernel(
        const unsigned short* __restrict__ Abf,   // [M][512]
        const unsigned short* __restrict__ Bt,    // [2048][512]
        const float* __restrict__ bias,           // [2048]
        unsigned short* __restrict__ u_bf,
        unsigned short* __restrict__ q_bf,
        unsigned short* __restrict__ k_bf,
        unsigned short* __restrict__ vt_bf) {
    const int K = E_;
    __shared__ __align__(16) unsigned short As[2][128 * 32];
    __shared__ __align__(16) unsigned short Bs[2][128 * 32];
    int tid = threadIdx.x;
    int w = tid >> 5, lane = tid & 31;
    int half = lane >> 4, l16 = lane & 15;
    int wm = w & 3, wn = w >> 2;
    int bm0 = blockIdx.x * 128;
    int bn0 = blockIdx.y * 128;
    unsigned asBase = (unsigned)(uintptr_t)&As[0][0];
    unsigned bsBase = (unsigned)(uintptr_t)&Bs[0][0];

    v8f zero = {};
    v8f acc[2][4];
#pragma unroll
    for (int i = 0; i < 2; ++i)
#pragma unroll
        for (int j = 0; j < 4; ++j) acc[i][j] = zero;

    tile_async_load(Abf, Bt, K, bm0, bn0, 0, tid, asBase, bsBase, 0);

    for (int k0 = 0; k0 < K; k0 += 32) {
        int cur = (k0 >> 5) & 1;
        wait_async0();
        __syncthreads();
        if (k0 + 32 < K)
            tile_async_load(Abf, Bt, K, bm0, bn0, k0 + 32, tid, asBase, bsBase, cur ^ 1);

        const unsigned short* Asc = &As[cur][0];
        const unsigned short* Bsc = &Bs[cur][0];
        BFrag a[2], b[4];
#pragma unroll
        for (int i = 0; i < 2; ++i) {
            int row = wm * 32 + i * 16 + l16;
            a[i].q[0] = *(const U16B*)&Asc[row * 32 + half * 8];
            a[i].q[1] = *(const U16B*)&Asc[row * 32 + 16 + half * 8];
        }
#pragma unroll
        for (int j = 0; j < 4; ++j) {
            int col = wn * 64 + j * 16 + l16;
            b[j].q[0] = *(const U16B*)&Bsc[col * 32 + half * 8];
            b[j].q[1] = *(const U16B*)&Bsc[col * 32 + 16 + half * 8];
        }
#pragma unroll
        for (int i = 0; i < 2; ++i)
#pragma unroll
            for (int j = 0; j < 4; ++j)
                acc[i][j] = wmma_bf16(a[i], b[j], acc[i][j]);
    }

    // Epilogue
#pragma unroll
    for (int i = 0; i < 2; ++i) {
#pragma unroll
        for (int j = 0; j < 4; ++j) {
#pragma unroll
            for (int r = 0; r < 8; ++r) {
                int row = bm0 + wm * 32 + i * 16 + half * 8 + r;   // token id
                int col = bn0 + wn * 64 + j * 16 + l16;
                float val = acc[i][j][r] + bias[col];
                int bb = row >> 11;          // / S_
                int ss = row & (S_ - 1);
                if (col < TH_) {
                    u_bf[(size_t)row * TH_ + col] = f2bf(silu(val));
                } else if (col < 2 * TH_) {
                    int cc = col - TH_;
                    int hh = cc >> 6, dd = cc & 63;
                    vt_bf[(((size_t)(bb * H_ + hh)) * HD_ + dd) * S_ + ss] = f2bf(val);
                } else if (col < 2 * TH_ + H_ * AD_) {
                    int cc = col - 2 * TH_;
                    int hh = cc >> 6, dd = cc & 63;
                    q_bf[(((size_t)(bb * H_ + hh)) * S_ + ss) * AD_ + dd] = f2bf(val);
                } else {
                    int cc = col - (2 * TH_ + H_ * AD_);
                    int hh = cc >> 6, dd = cc & 63;
                    k_bf[(((size_t)(bb * H_ + hh)) * S_ + ss) * AD_ + dd] = f2bf(val);
                }
            }
        }
    }
}

// ---------------------------------------------------------------------------
// Attention: per wave, a 16-query tile; causal loop over 32-key chunks.
// scores = silu(QK^T * alpha) * mask ; out = scores @ V
// ---------------------------------------------------------------------------
__global__ __launch_bounds__(256) void attn_kernel(
        const unsigned short* __restrict__ q_bf,   // [B*H][S][64]
        const unsigned short* __restrict__ k_bf,   // [B*H][S][64]
        const unsigned short* __restrict__ vt_bf,  // [B*H][64][S]
        const int* __restrict__ seq_lengths,
        unsigned short* __restrict__ attn_bf) {    // [M][512]
    __shared__ __align__(16) unsigned short ssc[8 * 16 * 32];  // per-wave score scratch
    int tid = threadIdx.x;
    int w = tid >> 5, lane = tid & 31;
    int half = lane >> 4, l16 = lane & 15;
    int bh = blockIdx.x;
    int b = bh >> 3, h = bh & 7;
    int qbase = blockIdx.y * 128 + w * 16;
    int slen = seq_lengths[b];
    const size_t headoff = (size_t)bh * S_ * 64;
    const unsigned short* Q  = q_bf + headoff;
    const unsigned short* Kp = k_bf + headoff;
    const unsigned short* Vt = vt_bf + headoff;
    unsigned short* myss = &ssc[w * 512];
    const float alpha = 0.125f;   // AD^-0.5

    // Q fragments: K-dim d in [0,32) and [32,64)
    BFrag aq[2];
    int qrow = qbase + l16;
#pragma unroll
    for (int kk = 0; kk < 2; ++kk) {
        aq[kk].q[0] = *(const U16B*)&Q[(size_t)qrow * 64 + kk * 32 + half * 8];
        aq[kk].q[1] = *(const U16B*)&Q[(size_t)qrow * 64 + kk * 32 + 16 + half * 8];
    }
    v8f zero = {};
    v8f acc[4];
#pragma unroll
    for (int dn = 0; dn < 4; ++dn) acc[dn] = zero;

    for (int m0 = 0; m0 <= qbase + 15; m0 += 32) {
        v8f sc[2];
        sc[0] = zero; sc[1] = zero;
#pragma unroll
        for (int c = 0; c < 2; ++c) {
            int key = m0 + c * 16 + l16;
#pragma unroll
            for (int kk = 0; kk < 2; ++kk) {
                BFrag kb;
                kb.q[0] = *(const U16B*)&Kp[(size_t)key * 64 + kk * 32 + half * 8];
                kb.q[1] = *(const U16B*)&Kp[(size_t)key * 64 + kk * 32 + 16 + half * 8];
                sc[c] = wmma_bf16(aq[kk], kb, sc[c]);
            }
        }
        // silu + mask in registers, then re-distribute via per-wave LDS scratch
#pragma unroll
        for (int c = 0; c < 2; ++c) {
#pragma unroll
            for (int r = 0; r < 8; ++r) {
                float v = sc[c][r] * alpha;
                float s = silu(v);
                int qq  = qbase + half * 8 + r;
                int key = m0 + c * 16 + l16;
                if (key > qq || key >= slen) s = 0.0f;
                myss[(half * 8 + r) * 32 + c * 16 + l16] = f2bf(s);
            }
        }
        BFrag sfrag;   // scores as A-matrix (16 rows x 32 keys)
        sfrag.q[0] = *(const U16B*)&myss[l16 * 32 + half * 8];
        sfrag.q[1] = *(const U16B*)&myss[l16 * 32 + 16 + half * 8];
#pragma unroll
        for (int dn = 0; dn < 4; ++dn) {
            BFrag vb;  // V as B-matrix (32 keys x 16 d), Vt is d-major
            int d = dn * 16 + l16;
            vb.q[0] = *(const U16B*)&Vt[(size_t)d * S_ + m0 + half * 8];
            vb.q[1] = *(const U16B*)&Vt[(size_t)d * S_ + m0 + 16 + half * 8];
            acc[dn] = wmma_bf16(sfrag, vb, acc[dn]);
        }
    }
    // store: attn[b][s][h*64+d]
#pragma unroll
    for (int dn = 0; dn < 4; ++dn) {
#pragma unroll
        for (int r = 0; r < 8; ++r) {
            int row = qbase + half * 8 + r;
            int col = h * 64 + dn * 16 + l16;
            attn_bf[((size_t)b * S_ + row) * TH_ + col] = f2bf(acc[dn][r]);
        }
    }
}

// ---------------------------------------------------------------------------
// LN2 + build y = [u, x, u * ln2(attn)] as bf16 [M][1536]
// ---------------------------------------------------------------------------
__global__ __launch_bounds__(256) void ln2_y_kernel(
        const unsigned short* __restrict__ attn_bf,
        const unsigned short* __restrict__ u_bf,
        const float* __restrict__ x,
        const float* __restrict__ g2, const float* __restrict__ b2,
        unsigned short* __restrict__ y_bf) {
    int w = threadIdx.x >> 5, lane = threadIdx.x & 31;
    int t = blockIdx.x * 8 + w;
    float vals[16];
    float s = 0.f, s2 = 0.f;
#pragma unroll
    for (int j = 0; j < 16; ++j) {
        float v = bf2f(attn_bf[(size_t)t * TH_ + lane + 32 * j]);
        vals[j] = v; s += v; s2 += v * v;
    }
#pragma unroll
    for (int off = 16; off >= 1; off >>= 1) {
        s  += __shfl_xor(s,  off, 32);
        s2 += __shfl_xor(s2, off, 32);
    }
    float mean = s * (1.0f / TH_);
    float var  = s2 * (1.0f / TH_) - mean * mean;
    float rinv = rsqrtf(var + 1e-6f);
#pragma unroll
    for (int j = 0; j < 16; ++j) {
        int i = lane + 32 * j;
        float nv = (vals[j] - mean) * rinv * g2[i] + b2[i];
        unsigned short ub = u_bf[(size_t)t * TH_ + i];
        float u = bf2f(ub);
        size_t base = (size_t)t * YK_;
        y_bf[base + i]              = ub;                          // u
        y_bf[base + TH_ + i]        = f2bf(x[(size_t)t * E_ + i]); // x
        y_bf[base + 2 * TH_ + i]    = f2bf(u * nv);                // u * norm_out
    }
}

// ---------------------------------------------------------------------------
// GEMM2: out = y(16384x1536) @ Wo(1536x512) + x   (f32 output)
// ---------------------------------------------------------------------------
__global__ __launch_bounds__(256) void gemm_out_kernel(
        const unsigned short* __restrict__ Ybf,   // [M][1536]
        const unsigned short* __restrict__ Bt,    // [512][1536]
        const float* __restrict__ x,
        float* __restrict__ out) {
    const int K = YK_;
    __shared__ __align__(16) unsigned short As[2][128 * 32];
    __shared__ __align__(16) unsigned short Bs[2][128 * 32];
    int tid = threadIdx.x;
    int w = tid >> 5, lane = tid & 31;
    int half = lane >> 4, l16 = lane & 15;
    int wm = w & 3, wn = w >> 2;
    int bm0 = blockIdx.x * 128;
    int bn0 = blockIdx.y * 128;
    unsigned asBase = (unsigned)(uintptr_t)&As[0][0];
    unsigned bsBase = (unsigned)(uintptr_t)&Bs[0][0];

    v8f zero = {};
    v8f acc[2][4];
#pragma unroll
    for (int i = 0; i < 2; ++i)
#pragma unroll
        for (int j = 0; j < 4; ++j) acc[i][j] = zero;

    tile_async_load(Ybf, Bt, K, bm0, bn0, 0, tid, asBase, bsBase, 0);

    for (int k0 = 0; k0 < K; k0 += 32) {
        int cur = (k0 >> 5) & 1;
        wait_async0();
        __syncthreads();
        if (k0 + 32 < K)
            tile_async_load(Ybf, Bt, K, bm0, bn0, k0 + 32, tid, asBase, bsBase, cur ^ 1);

        const unsigned short* Asc = &As[cur][0];
        const unsigned short* Bsc = &Bs[cur][0];
        BFrag a[2], b[4];
#pragma unroll
        for (int i = 0; i < 2; ++i) {
            int row = wm * 32 + i * 16 + l16;
            a[i].q[0] = *(const U16B*)&Asc[row * 32 + half * 8];
            a[i].q[1] = *(const U16B*)&Asc[row * 32 + 16 + half * 8];
        }
#pragma unroll
        for (int j = 0; j < 4; ++j) {
            int col = wn * 64 + j * 16 + l16;
            b[j].q[0] = *(const U16B*)&Bsc[col * 32 + half * 8];
            b[j].q[1] = *(const U16B*)&Bsc[col * 32 + 16 + half * 8];
        }
#pragma unroll
        for (int i = 0; i < 2; ++i)
#pragma unroll
            for (int j = 0; j < 4; ++j)
                acc[i][j] = wmma_bf16(a[i], b[j], acc[i][j]);
    }
#pragma unroll
    for (int i = 0; i < 2; ++i) {
#pragma unroll
        for (int j = 0; j < 4; ++j) {
#pragma unroll
            for (int r = 0; r < 8; ++r) {
                int row = bm0 + wm * 32 + i * 16 + half * 8 + r;
                int col = bn0 + wn * 64 + j * 16 + l16;
                size_t o = (size_t)row * E_ + col;
                out[o] = acc[i][j][r] + x[o];
            }
        }
    }
}

// ---------------------------------------------------------------------------
extern "C" void kernel_launch(void* const* d_in, const int* in_sizes, int n_in,
                              void* d_out, int out_size, void* d_ws, size_t ws_size,
                              hipStream_t stream) {
    const float* x      = (const float*)d_in[0];
    const int*   seqlen = (const int*)d_in[1];
    const float* wu     = (const float*)d_in[2];
    const float* bias   = (const float*)d_in[3];
    const float* wo     = (const float*)d_in[4];
    const float* g1     = (const float*)d_in[5];
    const float* b1     = (const float*)d_in[6];
    const float* g2     = (const float*)d_in[7];
    const float* b2     = (const float*)d_in[8];
    float* out = (float*)d_out;

    // Workspace carve-up (bf16 regions, all 256B aligned)
    char* ws = (char*)d_ws;
    size_t off = 0;
    auto carve = [&](size_t bytes) {
        char* p = ws + off;
        off += (bytes + 255) & ~(size_t)255;
        return p;
    };
    unsigned short* Xbf   = (unsigned short*)carve((size_t)M_ * E_ * 2);
    unsigned short* WtU   = (unsigned short*)carve((size_t)F_ * E_ * 2);
    unsigned short* WtO   = (unsigned short*)carve((size_t)E_ * YK_ * 2);
    unsigned short* Ubf   = (unsigned short*)carve((size_t)M_ * TH_ * 2);
    unsigned short* Qbf   = (unsigned short*)carve((size_t)B_ * H_ * S_ * AD_ * 2);
    unsigned short* Kbf   = (unsigned short*)carve((size_t)B_ * H_ * S_ * AD_ * 2);
    unsigned short* Vtbf  = (unsigned short*)carve((size_t)B_ * H_ * HD_ * S_ * 2);
    unsigned short* Attnb = (unsigned short*)carve((size_t)M_ * TH_ * 2);
    unsigned short* Ybf   = (unsigned short*)carve((size_t)M_ * YK_ * 2);

    convert_wu_kernel<<<dim3((E_ * F_) / 256), dim3(256), 0, stream>>>(wu, WtU);
    convert_wo_kernel<<<dim3((YK_ * E_) / 256), dim3(256), 0, stream>>>(wo, WtO);
    ln1_kernel<<<dim3(M_ / 8), dim3(256), 0, stream>>>(x, g1, b1, Xbf);
    gemm_uvqk_kernel<<<dim3(M_ / 128, F_ / 128), dim3(256), 0, stream>>>(
        Xbf, WtU, bias, Ubf, Qbf, Kbf, Vtbf);
    attn_kernel<<<dim3(B_ * H_, S_ / 128), dim3(256), 0, stream>>>(
        Qbf, Kbf, Vtbf, seqlen, Attnb);
    ln2_y_kernel<<<dim3(M_ / 8), dim3(256), 0, stream>>>(Attnb, Ubf, x, g2, b2, Ybf);
    gemm_out_kernel<<<dim3(M_ / 128, E_ / 128), dim3(256), 0, stream>>>(
        Ybf, WtO, x, out);
}
